// SetAbstraction_22531398435382
// MI455X (gfx1250) — matrix operand order repaired and checked by reading, not compiled, not run on hardware
//
#include <hip/hip_runtime.h>
#include <hip/hip_bf16.h>

// ---------------------------------------------------------------------------
// Set Abstraction (PointNet++-style) for MI455X / gfx1250.
//  Phase 1: FPS            (1 WG per batch, register-resident points)
//  Phase 2: weight swizzle (pre-pack W matrices into WMMA B-fragment layout, f16)
//  Phase 3: ball query     (per-thread nearest-32 insertion sort)
//  Phase 4: fused MLP      (v_wmma_f32_16x16x32_f16, f32 accum, LDS staging)
// ---------------------------------------------------------------------------

typedef __attribute__((ext_vector_type(16))) _Float16 v16h;
typedef __attribute__((ext_vector_type(8)))  _Float16 v8h;
typedef __attribute__((ext_vector_type(8)))  float    v8f;

union V16H { v16h v; v8h h[2]; };

constexpr int kB       = 8;
constexpr int kN       = 8192;
constexpr int kC       = 64;
constexpr int kH       = 128;
constexpr int kNPOINT  = 2048;
constexpr int kNSAMPLE = 32;
constexpr float kR2    = 0.4f * 0.4f;

// ---------------------------------------------------------------------------
// Phase 1: farthest point sampling. One workgroup per batch; each of 256
// threads owns 32 points in registers. Writes new_xyz directly to d_out.
// ---------------------------------------------------------------------------
__global__ __launch_bounds__(256) void fps_kernel(const float* __restrict__ xyz,
                                                  float* __restrict__ newxyz) {
  const int b = blockIdx.x;
  const int t = threadIdx.x;
  const float* X = xyz + (size_t)b * kN * 3;

  float px[32], py[32], pz[32], dmin[32];
  __shared__ float sv[256];
  __shared__ int   si[256];
  __shared__ float bc[3];

#pragma unroll
  for (int j = 0; j < 32; ++j) {
    int p = t + j * 256;
    px[j] = X[p * 3 + 0];
    py[j] = X[p * 3 + 1];
    pz[j] = X[p * 3 + 2];
  }
  // first centroid = point 0
  float fx = X[0], fy = X[1], fz = X[2];
#pragma unroll
  for (int j = 0; j < 32; ++j) {
    float dx = px[j] - fx, dy = py[j] - fy, dz = pz[j] - fz;
    dmin[j] = dx * dx + dy * dy + dz * dz;
  }
  if (t == 0) {
    float* o = newxyz + (size_t)(b * kNPOINT) * 3;
    o[0] = fx; o[1] = fy; o[2] = fz;
  }

  for (int it = 1; it < kNPOINT; ++it) {
    float best = -1.0f; int bi = 0;
#pragma unroll
    for (int j = 0; j < 32; ++j) {
      if (dmin[j] > best) { best = dmin[j]; bi = t + j * 256; }
    }
    sv[t] = best; si[t] = bi;
    __syncthreads();
    for (int s = 128; s > 0; s >>= 1) {
      if (t < s) {
        if (sv[t + s] > sv[t] || (sv[t + s] == sv[t] && si[t + s] < si[t])) {
          sv[t] = sv[t + s]; si[t] = si[t + s];
        }
      }
      __syncthreads();
    }
    if (t == 0) {
      int w = si[0];
      float wx = X[w * 3], wy = X[w * 3 + 1], wz = X[w * 3 + 2];
      bc[0] = wx; bc[1] = wy; bc[2] = wz;
      float* o = newxyz + (size_t)(b * kNPOINT + it) * 3;
      o[0] = wx; o[1] = wy; o[2] = wz;
    }
    __syncthreads();
    float wx = bc[0], wy = bc[1], wz = bc[2];
#pragma unroll
    for (int j = 0; j < 32; ++j) {
      float dx = px[j] - wx, dy = py[j] - wy, dz = pz[j] - wz;
      float d = dx * dx + dy * dy + dz * dz;
      dmin[j] = fminf(dmin[j], d);
    }
    __syncthreads();
  }
}

// ---------------------------------------------------------------------------
// Phase 2: swizzle a (K x 128) f32 weight matrix into WMMA B-fragment order,
// f16, zero-padded to KC*32 rows. Layout: [nt(8)][kc][lane(32)][half(16)].
// B-frag lane mapping (16x16x32 f16): N = lane&15, g = lane>>4, K = h<8 ? 8*g + h : 16 + 8*g + (h-8).
// ---------------------------------------------------------------------------
__global__ void swz_kernel(const float* __restrict__ W, _Float16* __restrict__ dst,
                           int Kreal, int KC) {
  int idx = blockIdx.x * 256 + threadIdx.x;
  int total = 8 * KC * 512;
  if (idx >= total) return;
  int h    = idx & 15;
  int lane = (idx >> 4) & 31;
  int kc   = (idx >> 9) % KC;
  int nt   = (idx >> 9) / KC;
  int g    = lane >> 4;
  int k    = kc * 32 + ((h < 8) ? (8 * g + h) : (16 + 8 * g + (h - 8)));
  int n    = nt * 16 + (lane & 15);
  float v  = (k < Kreal) ? W[k * kH + n] : 0.0f;
  dst[idx] = (_Float16)v;
}

// ---------------------------------------------------------------------------
// Phase 3: ball query / nearest-32. One thread per centroid; register-resident
// 32-entry insertion sort (stable, matches top_k lowest-index tie behavior).
// Out-of-radius distance -> 3e37 so pads fill in index order.
// ---------------------------------------------------------------------------
__global__ __launch_bounds__(256) void ballquery_kernel(const float* __restrict__ xyz,
                                                        const float* __restrict__ newxyz,
                                                        int* __restrict__ gidx) {
  int tid = blockIdx.x * 256 + threadIdx.x;
  if (tid >= kB * kNPOINT) return;
  int b = tid / kNPOINT;
  const float* X = xyz + (size_t)b * kN * 3;
  const float* q = newxyz + (size_t)tid * 3;
  float qx = q[0], qy = q[1], qz = q[2];

  float bd[32]; int bi[32];
#pragma unroll
  for (int j = 0; j < 32; ++j) { bd[j] = 3.4e38f; bi[j] = 0; }

  for (int i = 0; i < kN; ++i) {
    float dx = X[i * 3 + 0] - qx;
    float dy = X[i * 3 + 1] - qy;
    float dz = X[i * 3 + 2] - qz;
    float d = dx * dx + dy * dy + dz * dz;
    d = (d <= kR2) ? d : 3.0e37f;
    if (d < bd[31]) {
      float cd = d; int ci = i;
#pragma unroll
      for (int j = 0; j < 32; ++j) {
        if (cd < bd[j]) {
          float td = bd[j]; int ti = bi[j];
          bd[j] = cd; bi[j] = ci;
          cd = td; ci = ti;
        }
      }
    }
  }
  int* o = gidx + (size_t)tid * kNSAMPLE;
#pragma unroll
  for (int j = 0; j < 32; ++j) o[j] = bi[j];
}

// ---------------------------------------------------------------------------
// Phase 4: fused MLP. 128 threads = 4 waves per WG; 2 centroids (64 rows).
// Each wave owns a 16-row M-tile; A staged in LDS f16, B fragments streamed
// from the pre-swizzled (L2-resident) weight buffers.
// ---------------------------------------------------------------------------

// A-frag (16-bit A 16x32 layout): lane row = M=lane&15, g=lane>>4.
// Halves 0..7 -> K = 8g+0..7 ; halves 8..15 -> K = 16+8g+0..7  (two b128 runs).
__device__ __forceinline__ v16h load_a_frag(const _Float16* base, int stride,
                                            int row, int kc, int g) {
  V16H u;
  const _Float16* p = base + row * stride + kc * 32 + 8 * g;
  u.h[0] = *(const v8h*)(p);
  u.h[1] = *(const v8h*)(p + 16);
  return u.v;
}

__device__ __forceinline__ v16h load_b_frag(const _Float16* w, int nt, int KC,
                                            int kc, int lane) {
  V16H u;
  const v8h* p = (const v8h*)(w + (((size_t)(nt * KC + kc) * 32 + lane) << 4));
  u.h[0] = p[0];
  u.h[1] = p[1];
  return u.v;
}

// One layer: out[nt][v] = (A @ W)[M, nt*16 + (lane&15)] + bias
__device__ __forceinline__ void layer_gemm(const _Float16* Albase, int astride, int KC,
                                           const _Float16* Wswz,
                                           const float* __restrict__ bias,
                                           int mBase, int lane, float out[8][8]) {
  const int g   = lane >> 4;
  const int row = mBase + (lane & 15);
#pragma unroll
  for (int nt = 0; nt < 8; ++nt) {
    v8f c = {};
    for (int kc = 0; kc < KC; ++kc) {
      v16h a = load_a_frag(Albase, astride, row, kc, g);
      v16h b = load_b_frag(Wswz, nt, KC, kc, lane);
      c = __builtin_amdgcn_wmma_f32_16x16x32_f16(false, a, false, b,
                                                 (short)0, c, false, false);
    }
    float bv = bias[nt * 16 + (lane & 15)];
#pragma unroll
    for (int v = 0; v < 8; ++v) out[nt][v] = c[v] + bv;
  }
}

__global__ __launch_bounds__(128) void sa_mlp_kernel(
    const float* __restrict__ xyz, const float* __restrict__ feat,
    const float* __restrict__ newxyz, const int* __restrict__ gidx,
    const _Float16* __restrict__ w1a, const float* __restrict__ b1a,
    const _Float16* __restrict__ w1b, const float* __restrict__ b1b,
    const _Float16* __restrict__ ww1, const float* __restrict__ bw1,
    const _Float16* __restrict__ ww2, const float* __restrict__ bw2,
    float* __restrict__ fout_g) {
  __shared__ __align__(16) _Float16 AstaA[64][160];  // f_in (K96) / w_in (K160)
  __shared__ __align__(16) _Float16 AstaB[64][128];  // h / hw
  __shared__ float colsum[2][kH];
  __shared__ float foutl[2][kH];

  const int wg    = blockIdx.x;          // 0 .. B*NPOINT/2 - 1
  const int b     = wg >> 10;            // 1024 centroid-pairs per batch
  const int c0    = (wg & 1023) * 2;
  const int t     = threadIdx.x;
  const int lane  = t & 31;
  const int wv    = t >> 5;
  const int mBase = wv * 16;
  const int g     = lane >> 4;
  const int nlo   = lane & 15;

  // ---- stage 1: gather f_in = [gn, features] into AstaA (rows 0..63) ------
  if (t < 64) {
    int c  = c0 + (t >> 5);
    int s  = t & 31;
    int gi = gidx[((size_t)(b * kNPOINT + c)) * kNSAMPLE + s];
    const float* q = newxyz + (size_t)(b * kNPOINT + c) * 3;
    const float* p = xyz + (size_t)(b * kN + gi) * 3;
    float gx = p[0] - q[0], gy = p[1] - q[1], gz = p[2] - q[2];
    AstaA[t][0] = (_Float16)gx;
    AstaA[t][1] = (_Float16)gy;
    AstaA[t][2] = (_Float16)gz;
    const float* f = feat + (size_t)(b * kN + gi) * kC;
#pragma unroll 4
    for (int cc = 0; cc < kC; ++cc) AstaA[t][3 + cc] = (_Float16)f[cc];
#pragma unroll
    for (int cc = 67; cc < 96; ++cc) AstaA[t][cc] = (_Float16)0.0f;
  }
  ((float*)colsum)[t] = 0.0f; ((float*)colsum)[t + 128] = 0.0f;
  ((float*)foutl)[t]  = 0.0f; ((float*)foutl)[t + 128]  = 0.0f;
  __syncthreads();

  // ---- layer 1: relu(f_in @ W1a + b1a) -> AstaB ---------------------------
  {
    float h1[8][8];
    layer_gemm(&AstaA[0][0], 160, 3, w1a, b1a, mBase, lane, h1);
#pragma unroll
    for (int nt = 0; nt < 8; ++nt)
#pragma unroll
      for (int v = 0; v < 8; ++v) {
        float x = h1[nt][v]; x = x > 0.0f ? x : 0.0f;
        AstaB[mBase + 8 * g + v][nt * 16 + nlo] = (_Float16)x;
      }
  }
  __syncthreads();

  // ---- layer 2: f_prime = relu(h @ W1b + b1b); column sums for mean -------
  float fp[8][8];
  layer_gemm(&AstaB[0][0], 128, 4, w1b, b1b, mBase, lane, fp);
#pragma unroll
  for (int nt = 0; nt < 8; ++nt)
#pragma unroll
    for (int v = 0; v < 8; ++v) {
      float x = fp[nt][v]; x = x > 0.0f ? x : 0.0f;
      fp[nt][v] = x;
      int M = mBase + 8 * g + v;
      atomicAdd(&colsum[M >> 5][nt * 16 + nlo], x);
    }
  __syncthreads();

  // ---- stage: w_in = [gn, f_prime - mean] into AstaA ----------------------
#pragma unroll
  for (int nt = 0; nt < 8; ++nt)
#pragma unroll
    for (int v = 0; v < 8; ++v) {
      int M = mBase + 8 * g + v;
      int col = nt * 16 + nlo;
      float mean = colsum[M >> 5][col] * (1.0f / 32.0f);
      AstaA[M][3 + col] = (_Float16)(fp[nt][v] - mean);
    }
  if (t < 64) {
#pragma unroll
    for (int cc = 131; cc < 160; ++cc) AstaA[t][cc] = (_Float16)0.0f;
  }
  __syncthreads();

  // ---- layer 3: hw = relu(w_in @ Ww1 + bw1) -> AstaB ----------------------
  {
    float h3[8][8];
    layer_gemm(&AstaA[0][0], 160, 5, ww1, bw1, mBase, lane, h3);
#pragma unroll
    for (int nt = 0; nt < 8; ++nt)
#pragma unroll
      for (int v = 0; v < 8; ++v) {
        float x = h3[nt][v]; x = x > 0.0f ? x : 0.0f;
        AstaB[mBase + 8 * g + v][nt * 16 + nlo] = (_Float16)x;
      }
  }
  __syncthreads();

  // ---- layer 4 (fused): alpha = sigmoid(hw @ Ww2 + bw2); sum alpha*fp -----
  const int row = mBase + nlo;
#pragma unroll
  for (int nt = 0; nt < 8; ++nt) {
    v8f c = {};
    for (int kc = 0; kc < 4; ++kc) {
      v16h a = load_a_frag(&AstaB[0][0], 128, row, kc, g);
      v16h bfr = load_b_frag(ww2, nt, 4, kc, lane);
      c = __builtin_amdgcn_wmma_f32_16x16x32_f16(false, a, false, bfr,
                                                 (short)0, c, false, false);
    }
    float bv = bw2[nt * 16 + nlo];
#pragma unroll
    for (int v = 0; v < 8; ++v) {
      float alpha = 1.0f / (1.0f + __expf(-(c[v] + bv)));
      int M = mBase + 8 * g + v;
      atomicAdd(&foutl[M >> 5][nt * 16 + nlo], alpha * fp[nt][v]);
    }
  }
  __syncthreads();

  // ---- write f_out --------------------------------------------------------
#pragma unroll
  for (int rr = 0; rr < 2; ++rr) {
    int idx = t + rr * 128;
    int cc  = idx >> 7;
    int col = idx & 127;
    fout_g[((size_t)(b * kNPOINT + c0 + cc)) * kH + col] = foutl[cc][col];
  }
}

// ---------------------------------------------------------------------------
extern "C" void kernel_launch(void* const* d_in, const int* in_sizes, int n_in,
                              void* d_out, int out_size, void* d_ws, size_t ws_size,
                              hipStream_t stream) {
  const float* xyz  = (const float*)d_in[0];
  const float* feat = (const float*)d_in[1];
  const float* W1a  = (const float*)d_in[2];
  const float* b1a  = (const float*)d_in[3];
  const float* W1b  = (const float*)d_in[4];
  const float* b1b  = (const float*)d_in[5];
  const float* Ww1  = (const float*)d_in[6];
  const float* bw1  = (const float*)d_in[7];
  const float* Ww2  = (const float*)d_in[8];
  const float* bw2  = (const float*)d_in[9];

  float* out    = (float*)d_out;
  float* newxyz = out;                              // (B, NPOINT, 3)
  float* fout   = out + (size_t)kB * kNPOINT * 3;   // (B, NPOINT, H)

  char* ws = (char*)d_ws;
  int* gidx = (int*)ws;
  size_t off = (size_t)kB * kNPOINT * kNSAMPLE * sizeof(int);   // 2 MB
  _Float16* w1s = (_Float16*)(ws + off); off += (size_t)8 * 3 * 512 * 2;
  _Float16* w2s = (_Float16*)(ws + off); off += (size_t)8 * 4 * 512 * 2;
  _Float16* w3s = (_Float16*)(ws + off); off += (size_t)8 * 5 * 512 * 2;
  _Float16* w4s = (_Float16*)(ws + off); off += (size_t)8 * 4 * 512 * 2;

  fps_kernel<<<kB, 256, 0, stream>>>(xyz, newxyz);

  swz_kernel<<<(8 * 3 * 512 + 255) / 256, 256, 0, stream>>>(W1a, w1s, 67, 3);
  swz_kernel<<<(8 * 4 * 512 + 255) / 256, 256, 0, stream>>>(W1b, w2s, 128, 4);
  swz_kernel<<<(8 * 5 * 512 + 255) / 256, 256, 0, stream>>>(Ww1, w3s, 131, 5);
  swz_kernel<<<(8 * 4 * 512 + 255) / 256, 256, 0, stream>>>(Ww2, w4s, 128, 4);

  ballquery_kernel<<<(kB * kNPOINT + 255) / 256, 256, 0, stream>>>(xyz, newxyz, gidx);

  sa_mlp_kernel<<<kB * kNPOINT / 2, 128, 0, stream>>>(
      xyz, feat, newxyz, gidx, w1s, b1a, w2s, b1b, w3s, bw1, w4s, bw2, fout);
}
